// MultiHeadSelfAttention_25323127177224
// MI455X (gfx1250) — compile-verified, matrix operand order checked by reading, not run on hardware
//
#include <hip/hip_runtime.h>

typedef __attribute__((ext_vector_type(16))) __bf16 v16bf;
typedef __attribute__((ext_vector_type(8)))  float  v8f;
typedef __attribute__((__vector_size__(4 * sizeof(int)))) int v4i;

#define B_  4
#define S_  2048
#define D_  1024
#define H_  16
#define HD_ 64
#define M_  (B_ * S_)   // 8192

// ---------------------------------------------------------------------------
// Async global->LDS copy path (CDNA5 GLOBAL_LOAD_ASYNC_TO_LDS_B128, ASYNCcnt).
// Builtin exists on this toolchain; params are v4i pointers. Cast global src
// to as(1) v4i* and LDS dst to as(3) v4i* (implicitly decays to generic if
// the builtin's params are unqualified).
// ---------------------------------------------------------------------------
#if __has_builtin(__builtin_amdgcn_global_load_async_to_lds_b128) && \
    __has_builtin(__builtin_amdgcn_s_wait_asynccnt)
#define ASYNC_CP(g, l)                                                       \
    __builtin_amdgcn_global_load_async_to_lds_b128(                          \
        (__attribute__((address_space(1))) v4i*)(g),                         \
        (__attribute__((address_space(3))) v4i*)(l), 0, 0)
#define ASYNC_WAIT() __builtin_amdgcn_s_wait_asynccnt(0)
#else
#define ASYNC_CP(g, l) (*(uint4*)(l) = *(const uint4*)(g))
#define ASYNC_WAIT() ((void)0)
#endif

// ---------------------------------------------------------------------------
// WMMA helper: D = A(16x32 bf16) x B(32x16 bf16) + C (f32)
// ---------------------------------------------------------------------------
__device__ inline v8f wmma_bf16(v16bf a, v16bf b, v8f c) {
    return __builtin_amdgcn_wmma_f32_16x16x32_bf16(
        false, a, false, b, (short)0, c, false, false);
}

// ---------------------------------------------------------------------------
// Fragment loaders (wave32, 16-bit 16x32 A / 32x16 B layouts, ISA 7.12.2).
// Per lane: two contiguous 16-byte chunks at element offsets half*8 and
// 16 + half*8.  Caller passes rowbase already including (lane&15)*ld.
// Works for global (global_load_b128) and LDS (ds_load_b128) pointers.
// ---------------------------------------------------------------------------
__device__ inline v16bf frag_bf16(const __bf16* rowbase) {
    const int hl = (threadIdx.x >> 4) & 1;
    union { v16bf v; uint4 q[2]; } u;
    u.q[0] = *reinterpret_cast<const uint4*>(rowbase + hl * 8);
    u.q[1] = *reinterpret_cast<const uint4*>(rowbase + 16 + hl * 8);
    return u.v;
}

__device__ inline v16bf frag_f32(const float* rowbase) {
    const int hl = (threadIdx.x >> 4) & 1;
    const float4* p0 = reinterpret_cast<const float4*>(rowbase + hl * 8);
    const float4* p1 = reinterpret_cast<const float4*>(rowbase + 16 + hl * 8);
    float4 f0 = p0[0], f1 = p0[1], f2 = p1[0], f3 = p1[1];
    v16bf v;
    v[0]  = (__bf16)f0.x; v[1]  = (__bf16)f0.y; v[2]  = (__bf16)f0.z; v[3]  = (__bf16)f0.w;
    v[4]  = (__bf16)f1.x; v[5]  = (__bf16)f1.y; v[6]  = (__bf16)f1.z; v[7]  = (__bf16)f1.w;
    v[8]  = (__bf16)f2.x; v[9]  = (__bf16)f2.y; v[10] = (__bf16)f2.z; v[11] = (__bf16)f2.w;
    v[12] = (__bf16)f3.x; v[13] = (__bf16)f3.y; v[14] = (__bf16)f3.z; v[15] = (__bf16)f3.w;
    return v;
}

// ---------------------------------------------------------------------------
// Kernel 1: weight convert + transpose  Wt[n][k] = (bf16) W[k][n]
// ---------------------------------------------------------------------------
__global__ __launch_bounds__(256)
void wt_convert_kernel(const float* __restrict__ W, __bf16* __restrict__ Wt) {
    int idx = blockIdx.x * 256 + threadIdx.x;      // over 1024*1024
    int k = idx >> 10;
    int n = idx & 1023;
    Wt[(size_t)n * 1024 + k] = (__bf16)W[idx];
}

// ---------------------------------------------------------------------------
// Cooperative B-tile stage: Wt rows [blockN, blockN+128) x 32 K-elements into
// LDS (row stride 40 elems = 80B, bank-conflict-free for b128 frag reads).
// 512 x 16B chunks, 2 per thread (256 threads).
// ---------------------------------------------------------------------------
__device__ inline void stage_b(const __bf16* __restrict__ Wt, int blockN, int k0,
                               __bf16* lds) {
    const int t = threadIdx.x;
#pragma unroll
    for (int i = 0; i < 2; ++i) {
        int c = t + i * 256;
        int row = c >> 2, sub = c & 3;
        ASYNC_CP(Wt + (size_t)(blockN + row) * 1024 + k0 + sub * 8,
                 lds + row * 40 + sub * 8);
    }
}

// ---------------------------------------------------------------------------
// Kernel 2: projection GEMM  C = A(f32)[M,1024] x Wt^T + bias  -> bf16
//   mode 0: out[b][h][s][hd]   (Q, K)     mode 1: out[b][h][hd][s]   (V^T)
// Block 256 thr (8 waves: 4 M-tiles x 2 N-groups), tile 64x128, LDS-staged B.
// ---------------------------------------------------------------------------
__global__ __launch_bounds__(256)
void proj_gemm_kernel(const float* __restrict__ A, const __bf16* __restrict__ Wt,
                      const float* __restrict__ bias, __bf16* __restrict__ out,
                      int mode) {
    __shared__ alignas(16) __bf16 ldsB[2][128 * 40];
    const int lane = threadIdx.x & 31;
    const int wave = threadIdx.x >> 5;
    const int hl   = lane >> 4;
    const int n0   = lane & 15;
    const int rowBase = blockIdx.x * 64 + (wave & 3) * 16;
    const int blockN  = blockIdx.y * 128;
    const int colBase = blockN + (wave >> 2) * 64;
    const int row = rowBase + n0;

    stage_b(Wt, blockN, 0, ldsB[0]);
    int cur = 0;
    v8f acc[4] = {};
    for (int k0 = 0; k0 < 1024; k0 += 32) {
        ASYNC_WAIT();
        __syncthreads();
        if (k0 + 32 < 1024) stage_b(Wt, blockN, k0 + 32, ldsB[cur ^ 1]);
        __builtin_prefetch((const void*)(A + (size_t)row * 1024 + k0 + 32), 0, 1);
        v16bf a = frag_f32(A + (size_t)row * 1024 + k0);
#pragma unroll
        for (int j = 0; j < 4; ++j) {
            v16bf b = frag_bf16(ldsB[cur] + ((wave >> 2) * 64 + j * 16 + n0) * 40);
            acc[j] = wmma_bf16(a, b, acc[j]);
        }
        __syncthreads();
        cur ^= 1;
    }
#pragma unroll
    for (int j = 0; j < 4; ++j) {
        int n  = colBase + j * 16 + n0;
        float bv = bias[n];
        int h  = n >> 6, hd = n & 63;
#pragma unroll
        for (int r = 0; r < 8; ++r) {
            int m  = rowBase + r + 8 * hl;
            int bb = m >> 11;                 // m / S
            int s  = m & (S_ - 1);
            float val = acc[j][r] + bv;
            size_t idx = (mode == 0)
                ? ((size_t)(bb * H_ + h) * S_ + s) * HD_ + hd
                : ((size_t)(bb * H_ + h) * HD_ + hd) * S_ + s;
            out[idx] = (__bf16)val;
        }
    }
}

// ---------------------------------------------------------------------------
// Kernel 3: flash attention.  Block = 256 thr (8 waves) shares one (b,h) and
// 128 consecutive queries; K/V 32-key tiles staged to LDS (async, double-buf).
// Qp/Kp: [B,H,S,HD]  VpT: [B,H,HD,S]  attn: [B,S,D]  (all bf16)
// ---------------------------------------------------------------------------
__global__ __launch_bounds__(256)
void attn_kernel(const __bf16* __restrict__ Qp, const __bf16* __restrict__ Kp,
                 const __bf16* __restrict__ VpT, __bf16* __restrict__ attn) {
    __shared__ alignas(16) __bf16 ldsK[2][32 * 72];  // key row stride 72
    __shared__ alignas(16) __bf16 ldsV[2][64 * 40];  // hd  row stride 40
    __shared__ alignas(16) __bf16 pbuf[8][16 * 40];  // per-wave P tile

    const int lane = threadIdx.x & 31;
    const int wave = threadIdx.x >> 5;
    const int hl   = lane >> 4;
    const int n0   = lane & 15;
    const int bh   = blockIdx.x >> 4;                // b*H + h
    const int qt   = (blockIdx.x & 15) * 8 + wave;   // 16-query tile 0..127
    const float scale = 0.125f;                      // 1/sqrt(HD)

    const __bf16* Kbase = Kp  + (size_t)bh * S_ * HD_;
    const __bf16* Vbase = VpT + (size_t)bh * HD_ * S_;

    const __bf16* Qb = Qp + ((size_t)bh * S_ + qt * 16 + n0) * HD_;
    v16bf aq0 = frag_bf16(Qb);
    v16bf aq1 = frag_bf16(Qb + 32);

    float mrun[8], lrun[8];
    v8f o[4] = {};
#pragma unroll
    for (int r = 0; r < 8; ++r) { mrun[r] = -INFINITY; lrun[r] = 0.0f; }

    // cooperative stage of one 32-key K tile (32x64) + V tile (64x32):
    // 256+256 16B chunks, 2 per thread.
    auto stage_kv = [&](int kt, int buf) {
        const int t = threadIdx.x;
        {   int r = t >> 3, sub = t & 7;             // K: 32 rows x 8 chunks
            ASYNC_CP(Kbase + (size_t)(kt + r) * HD_ + sub * 8,
                     ldsK[buf] + r * 72 + sub * 8);
        }
        {   int r = t >> 2, sub = t & 3;             // V: 64 rows x 4 chunks
            ASYNC_CP(Vbase + (size_t)r * S_ + kt + sub * 8,
                     ldsV[buf] + r * 40 + sub * 8);
        }
    };

    stage_kv(0, 0);
    int cur = 0;
    for (int kt = 0; kt < S_; kt += 32) {
        ASYNC_WAIT();
        __syncthreads();
        if (kt + 32 < S_) stage_kv(kt + 32, cur ^ 1);

        const __bf16* kb0 = ldsK[cur] + n0 * 72;
        const __bf16* kb1 = ldsK[cur] + (16 + n0) * 72;
        v8f s0 = {}, s1 = {};
        s0 = wmma_bf16(aq0, frag_bf16(kb0),      s0);
        s0 = wmma_bf16(aq1, frag_bf16(kb0 + 32), s0);
        s1 = wmma_bf16(aq0, frag_bf16(kb1),      s1);
        s1 = wmma_bf16(aq1, frag_bf16(kb1 + 32), s1);

#pragma unroll
        for (int r = 0; r < 8; ++r) {
            float t0 = s0[r] * scale, t1 = s1[r] * scale;
            float mx = fmaxf(t0, t1);
            for (int d = 1; d < 16; d <<= 1) mx = fmaxf(mx, __shfl_xor(mx, d, 32));
            float mnew  = fmaxf(mrun[r], mx);
            float alpha = __expf(mrun[r] - mnew);
            float p0 = __expf(t0 - mnew), p1 = __expf(t1 - mnew);
            float ls = p0 + p1;
            for (int d = 1; d < 16; d <<= 1) ls += __shfl_xor(ls, d, 32);
            lrun[r] = lrun[r] * alpha + ls;
            mrun[r] = mnew;
#pragma unroll
            for (int j = 0; j < 4; ++j) o[j][r] *= alpha;
            pbuf[wave][(r + 8 * hl) * 40 + n0]      = (__bf16)p0;
            pbuf[wave][(r + 8 * hl) * 40 + 16 + n0] = (__bf16)p1;
        }
        // LDS ops are in-order within a wave: pbuf RAW needs no barrier.
        v16bf ap = frag_bf16(&pbuf[wave][n0 * 40]);
#pragma unroll
        for (int j = 0; j < 4; ++j) {
            v16bf bv = frag_bf16(ldsV[cur] + (j * 16 + n0) * 40);
            o[j] = wmma_bf16(ap, bv, o[j]);
        }
        __syncthreads();
        cur ^= 1;
    }

    const int bb = bh >> 4, h = bh & 15;
#pragma unroll
    for (int r = 0; r < 8; ++r) {
        float inv = 1.0f / lrun[r];
        int s = qt * 16 + r + 8 * hl;
#pragma unroll
        for (int j = 0; j < 4; ++j)
            attn[((size_t)bb * S_ + s) * D_ + h * HD_ + j * 16 + n0] =
                (__bf16)(o[j][r] * inv);
    }
}

// ---------------------------------------------------------------------------
// Kernel 4: output GEMM  out = attn(bf16)[M,1024] x Wt^T + bias -> f32
// ---------------------------------------------------------------------------
__global__ __launch_bounds__(256)
void out_gemm_kernel(const __bf16* __restrict__ A, const __bf16* __restrict__ Wt,
                     const float* __restrict__ bias, float* __restrict__ out) {
    __shared__ alignas(16) __bf16 ldsB[2][128 * 40];
    const int lane = threadIdx.x & 31;
    const int wave = threadIdx.x >> 5;
    const int hl   = lane >> 4;
    const int n0   = lane & 15;
    const int rowBase = blockIdx.x * 64 + (wave & 3) * 16;
    const int blockN  = blockIdx.y * 128;
    const int colBase = blockN + (wave >> 2) * 64;
    const int row = rowBase + n0;

    stage_b(Wt, blockN, 0, ldsB[0]);
    int cur = 0;
    v8f acc[4] = {};
    for (int k0 = 0; k0 < 1024; k0 += 32) {
        ASYNC_WAIT();
        __syncthreads();
        if (k0 + 32 < 1024) stage_b(Wt, blockN, k0 + 32, ldsB[cur ^ 1]);
        __builtin_prefetch((const void*)(A + (size_t)row * 1024 + k0 + 32), 0, 1);
        v16bf a = frag_bf16(A + (size_t)row * 1024 + k0);
#pragma unroll
        for (int j = 0; j < 4; ++j) {
            v16bf b = frag_bf16(ldsB[cur] + ((wave >> 2) * 64 + j * 16 + n0) * 40);
            acc[j] = wmma_bf16(a, b, acc[j]);
        }
        __syncthreads();
        cur ^= 1;
    }
#pragma unroll
    for (int j = 0; j < 4; ++j) {
        int n = colBase + j * 16 + n0;
        float bv = bias[n];
#pragma unroll
        for (int r = 0; r < 8; ++r) {
            int m = rowBase + r + 8 * hl;
            out[(size_t)m * 1024 + n] = acc[j][r] + bv;
        }
    }
}

// ---------------------------------------------------------------------------
extern "C" void kernel_launch(void* const* d_in, const int* in_sizes, int n_in,
                              void* d_out, int out_size, void* d_ws, size_t ws_size,
                              hipStream_t stream) {
    const float* q     = (const float*)d_in[0];
    const float* k     = (const float*)d_in[1];
    const float* v     = (const float*)d_in[2];
    const float* q_w   = (const float*)d_in[3];
    const float* q_b   = (const float*)d_in[4];
    const float* k_w   = (const float*)d_in[5];
    const float* k_b   = (const float*)d_in[6];
    const float* out_w = (const float*)d_in[7];
    const float* out_b = (const float*)d_in[8];
    float* out = (float*)d_out;

    __bf16* WtQ  = (__bf16*)d_ws;                       // [1024,1024]
    __bf16* WtK  = WtQ + (size_t)1024 * 1024;
    __bf16* WtO  = WtK + (size_t)1024 * 1024;
    __bf16* Qp   = WtO + (size_t)1024 * 1024;           // [B,H,S,HD]
    __bf16* Kp   = Qp  + (size_t)M_ * D_;
    __bf16* VpT  = Kp  + (size_t)M_ * D_;               // [B,H,HD,S]
    __bf16* attn = VpT + (size_t)M_ * D_;               // [B,S,D]

    // 1) weight convert + transpose
    wt_convert_kernel<<<4096, 256, 0, stream>>>(q_w,   WtQ);
    wt_convert_kernel<<<4096, 256, 0, stream>>>(k_w,   WtK);
    wt_convert_kernel<<<4096, 256, 0, stream>>>(out_w, WtO);

    // 2) projections (note reference bug: V uses q_w / q_b)
    dim3 ggrid(M_ / 64, D_ / 128);
    proj_gemm_kernel<<<ggrid, 256, 0, stream>>>(q, WtQ, q_b, Qp,  0);
    proj_gemm_kernel<<<ggrid, 256, 0, stream>>>(k, WtK, k_b, Kp,  0);
    proj_gemm_kernel<<<ggrid, 256, 0, stream>>>(v, WtQ, q_b, VpT, 1);

    // 3) flash attention: (B*H) x (S/128) blocks, 8 query-tiles per block
    attn_kernel<<<B_ * H_ * (S_ / 128), 256, 0, stream>>>(Qp, Kp, VpT, attn);

    // 4) output projection
    out_gemm_kernel<<<ggrid, 256, 0, stream>>>(attn, WtO, out_b, out);
}
